// EncoderDecoder_8435315770094
// MI455X (gfx1250) — matrix operand order repaired and checked by reading, not compiled, not run on hardware
//
#include <hip/hip_runtime.h>
#include <cstdint>
#include <cstddef>

// ---------------------------------------------------------------------------
// Types for CDNA5 (gfx1250) WMMA: wave32, v_wmma_f32_16x16x32_bf16
// ---------------------------------------------------------------------------
typedef __bf16 bf16;
typedef __attribute__((ext_vector_type(16))) __bf16 v16bf;
typedef __attribute__((ext_vector_type(8)))  __bf16 v8bf;
typedef __attribute__((ext_vector_type(8)))  float  v8f;
typedef __attribute__((ext_vector_type(4)))  uint32_t u32x4;
typedef __attribute__((ext_vector_type(4)))  int      i32x4;
typedef __attribute__((ext_vector_type(8)))  int      i32x8;

#define DEV static __device__ __forceinline__

DEV bf16 f2bf(float f) {
  union { float f; uint32_t u; } a; a.f = f;
  uint32_t r = a.u + 0x7FFFu + ((a.u >> 16) & 1u);   // round-to-nearest-even
  union { uint16_t u; bf16 b; } c; c.u = (uint16_t)(r >> 16);
  return c.b;
}
DEV float bf2f(bf16 b) {
  union { uint16_t u; bf16 b; } c; c.b = b;
  union { uint32_t u; float f; } a; a.u = ((uint32_t)c.u) << 16;
  return a.f;
}
DEV float sigm(float x) { return 1.0f / (1.0f + __expf(-x)); }

// A-fragment: 16x32 bf16 tile from row-major X[ld]; per ISA layout lane holds
// row (lane&15), K = base..base+7 and 16+base..16+base+7 (base = 8*(lane>>4)).
DEV v16bf frag_ld_a(const bf16* base, int ld, int r0, int k0) {
  int lane = threadIdx.x & 31;
  const bf16* q = base + (size_t)(r0 + (lane & 15)) * (size_t)ld
                       + (size_t)(k0 + ((lane >> 4) << 3));
  union { v16bf v; v8bf h[2]; } u;
  u.h[0] = *(const v8bf*)(q);        // 16B load
  u.h[1] = *(const v8bf*)(q + 16);   // 16B load
  return u.v;
}
// B-fragment: 32x16 bf16 tile, B^T supplied as row-major W[N,K]; lane holds
// column (lane&15), K = base..base+7 and 16+base..16+base+7 (base = 8*(lane>>4)).
DEV v16bf frag_ld_b(const bf16* base, int ld, int n0, int k0) {
  int lane = threadIdx.x & 31;
  const bf16* q = base + (size_t)(n0 + (lane & 15)) * (size_t)ld
                       + (size_t)(k0 + ((lane >> 4) << 3));
  union { v16bf v; v8bf h[2]; } u;
  u.h[0] = *(const v8bf*)(q);
  u.h[1] = *(const v8bf*)(q + 16);
  return u.v;
}
DEV v8f wmma_bf(v16bf a, v16bf b, v8f c) {
  return __builtin_amdgcn_wmma_f32_16x16x32_bf16(false, a, false, b, (short)0, c,
                                                 false, false);
}
// C/D 16x16 f32: lane -> N = lane&15, M = 8*(lane>>4)+v
DEV void frag_st(float* base, int ld, int m0, int n0, v8f acc, float bias) {
  int lane = threadIdx.x & 31;
  int n = n0 + (lane & 15);
  int mb = m0 + ((lane >> 4) << 3);
#pragma unroll
  for (int v = 0; v < 8; ++v) base[(size_t)(mb + v) * ld + n] = acc[v] + bias;
}
DEV void frag_acc(float* base, int ld, int m0, int n0, v8f acc, float bias) {
  int lane = threadIdx.x & 31;
  int n = n0 + (lane & 15);
  int mb = m0 + ((lane >> 4) << 3);
#pragma unroll
  for (int v = 0; v < 8; ++v) base[(size_t)(mb + v) * ld + n] += acc[v] + bias;
}

// ---------------------------------------------------------------------------
// Problem constants
// ---------------------------------------------------------------------------
#define SLEN 64
#define BSZ  64
#define TLEN 32
#define EDIM 300
#define EPAD 320
#define UDIM 256
#define VOCAB 32000
#define KD   832    // pad(300+512)
#define KO   1088   // pad(256+512+300)

// ---------------------------------------------------------------------------
// f32 -> bf16 weight conversion with optional column slice + K padding
// ---------------------------------------------------------------------------
__global__ void k_cvt(bf16* dst, const float* src, int N, int srcld, int col0,
                      int K, int Kp) {
  int i = blockIdx.x * blockDim.x + threadIdx.x;
  if (i >= N * Kp) return;
  int n = i / Kp, k = i - n * Kp;
  dst[i] = f2bf(k < K ? src[(size_t)n * srcld + col0 + k] : 0.0f);
}

// embedding gather -> bf16, padded rows
__global__ void k_gather(bf16* dst, const float* emb, const int* tok, int rows,
                         int E, int Kp) {
  int i = blockIdx.x * blockDim.x + threadIdx.x;
  if (i >= rows * Kp) return;
  int r = i / Kp, k = i - r * Kp;
  dst[i] = f2bf(k < E ? emb[(size_t)tok[r] * E + k] : 0.0f);
}

// ---------------------------------------------------------------------------
// Generic tiled GEMM: C[M,N] = X[M,K] * W[N,K]^T + bias[N]; 1 tile / wave
// ---------------------------------------------------------------------------
__global__ __launch_bounds__(256) void k_gemm(const bf16* __restrict__ X, int ldx,
                                              const bf16* __restrict__ W, int ldw,
                                              const float* __restrict__ bias,
                                              float* __restrict__ C, int ldc,
                                              int Mtiles, int Ntiles, int Ksteps) {
  int wid = (blockIdx.x * 256 + threadIdx.x) >> 5;
  if (wid >= Mtiles * Ntiles) return;
  int mt = wid % Mtiles, nt = wid / Mtiles;
  v8f acc = {};
  for (int k = 0; k < Ksteps; ++k) {
    v16bf a = frag_ld_a(X, ldx, mt * 16, k * 32);
    v16bf b = frag_ld_b(W, ldw, nt * 16, k * 32);
    acc = wmma_bf(a, b, acc);
  }
  float bv = bias ? bias[nt * 16 + (threadIdx.x & 15)] : 0.0f;
  frag_st(C, ldc, mt * 16, nt * 16, acc, bv);
}

// ---------------------------------------------------------------------------
// Sequential encoder GRU (one direction), 1 block / 16 waves, state in LDS.
// gi[s,b,768] precomputed (= x@W_ih^T + b_ih).  enc_bf layout: [b][s][512].
// ---------------------------------------------------------------------------
__global__ __launch_bounds__(512) void k_enc_gru(const float* __restrict__ gi,
                                                 const bf16* __restrict__ Whh,
                                                 const float* __restrict__ bhh,
                                                 bf16* __restrict__ enc_bf,
                                                 bf16* __restrict__ hfin, int dir) {
  __shared__ __align__(16) float G[BSZ * 768];      // 192 KB: gate pre-acts
  __shared__ __align__(16) float hf[BSZ * UDIM];    //  64 KB: h (f32)
  __shared__ __align__(16) bf16  hb[BSZ * UDIM];    //  32 KB: h (bf16) for WMMA
  int tid = threadIdx.x, wid = tid >> 5;
  for (int i = tid; i < BSZ * UDIM; i += 512) { hf[i] = 0.0f; hb[i] = f2bf(0.0f); }
  __syncthreads();
  for (int step = 0; step < SLEN; ++step) {
    int s = dir ? (SLEN - 1 - step) : step;
    // G = h @ Whh^T : 4 x 48 tiles, K=256 (8 k-steps); A from LDS (ds_load)
    for (int t = wid; t < 192; t += 16) {
      int mt = t & 3, nt = t >> 2;
      v8f acc = {};
#pragma unroll
      for (int k = 0; k < 8; ++k)
        acc = wmma_bf(frag_ld_a(hb, UDIM, mt * 16, k * 32),
                      frag_ld_b(Whh, UDIM, nt * 16, k * 32), acc);
      frag_st(G, 768, mt * 16, nt * 16, acc, 0.0f);
    }
    __syncthreads();
    const float* gis = gi + (size_t)s * BSZ * 768;
    for (int i = tid; i < BSZ * UDIM; i += 512) {
      int b = i >> 8, u = i & 255;
      const float* gb = gis + b * 768;
      float* Gb = G + b * 768;
      float r = sigm(gb[u]        + Gb[u]        + bhh[u]);
      float z = sigm(gb[256 + u]  + Gb[256 + u]  + bhh[256 + u]);
      float n = tanhf(gb[512 + u] + r * (Gb[512 + u] + bhh[512 + u]));
      float h = (1.0f - z) * n + z * hf[i];
      hf[i] = h;
      bf16 hv = f2bf(h);
      hb[i] = hv;
      enc_bf[(size_t)b * (SLEN * 512) + (size_t)s * 512 + dir * 256 + u] = hv;
      hfin[b * 512 + dir * 256 + u] = hv;   // final h survives last overwrite
    }
    __syncthreads();
  }
}

// state = tanh([h_f|h_b] @ W_enc_fc^T + b) ; writes decoder h (f32 + bf16)
__global__ __launch_bounds__(256) void k_state(const bf16* __restrict__ hfin,
                                               const bf16* __restrict__ Wfc,
                                               const float* __restrict__ bfc,
                                               float* __restrict__ hf32,
                                               bf16* __restrict__ hbf) {
  int wid = threadIdx.x >> 5, lane = threadIdx.x & 31;
  for (int t = wid; t < 64; t += 8) {
    int mt = t & 3, nt = t >> 2;
    v8f acc = {};
    for (int k = 0; k < 16; ++k)
      acc = wmma_bf(frag_ld_a(hfin, 512, mt * 16, k * 32),
                    frag_ld_b(Wfc, 512, nt * 16, k * 32), acc);
    int n = nt * 16 + (lane & 15);
    int mb = mt * 16 + ((lane >> 4) << 3);
    float bv = bfc[n];
#pragma unroll
    for (int v = 0; v < 8; ++v) {
      float val = tanhf(acc[v] + bv);
      hf32[(mb + v) * UDIM + n] = val;
      hbf[(mb + v) * UDIM + n] = f2bf(val);
    }
  }
}

// zero outputs[0] and the zero-pad tail of x_out (done once)
__global__ void k_init(float* out0, bf16* xout) {
  int i = blockIdx.x * blockDim.x + threadIdx.x;
  if (i < BSZ * VOCAB) out0[i] = 0.0f;
  if (i < BSZ * (KO - 1068)) {
    int b = i / (KO - 1068), k = i - b * (KO - 1068);
    xout[b * KO + 1068 + k] = f2bf(0.0f);
  }
}

// ---------------------------------------------------------------------------
// One decoder step: attention + context + GRU cell.  1 block, 16 waves.
// enc_proj = enc@W_attn_e^T + b_attn precomputed.  Produces x_out=[h2|ctx|emb].
// ---------------------------------------------------------------------------
__global__ __launch_bounds__(512) void k_dec_step(
    float* __restrict__ hf32, bf16* __restrict__ hbf,
    const bf16* __restrict__ Wah, const float* __restrict__ enc_proj,
    const bf16* __restrict__ enc_bf, const bf16* __restrict__ embt,
    const bf16* __restrict__ Wihd, const float* __restrict__ bihd,
    const bf16* __restrict__ Whhd, const float* __restrict__ bhhd,
    bf16* __restrict__ xout) {
  __shared__ __align__(16) char pool[303104];          // 296 KB of the 320 KB WGP LDS
  float* hp = (float*)pool;                            // [64,256] h-projection
  float* sc = (float*)(pool + 65536);                  // [64,64] scores
  float* wg = (float*)(pool + 81920);                  // [64,64] softmax weights
  float* G  = (float*)pool;                            // [64,768] gates (reuses hp/sc/wg)
  float* Gn = (float*)(pool + 196608);                 // [64,256] h_n part (reuses xd)
  bf16*  xd = (bf16*)(pool + 196608);                  // [64,832] = [embt|ctx]
  int tid = threadIdx.x, lane = tid & 31, wid = tid >> 5;

  // P0: hp = h @ W_attn_h^T  (no bias; b_attn folded in enc_proj)
  for (int t = wid; t < 64; t += 16) {
    int mt = t & 3, nt = t >> 2;
    v8f acc = {};
#pragma unroll
    for (int k = 0; k < 8; ++k)
      acc = wmma_bf(frag_ld_a(hbf, UDIM, mt * 16, k * 32),
                    frag_ld_b(Wah, UDIM, nt * 16, k * 32), acc);
    frag_st(hp, UDIM, mt * 16, nt * 16, acc, 0.0f);
  }
  __syncthreads();
  // P1: scores[b,s] = sum_a tanh(hp[b,a] + enc_proj[b,s,a])
  for (int i = tid; i < BSZ * SLEN; i += 512) {
    int b = i >> 6, s = i & 63;
    const float* ep = enc_proj + (size_t)(b * SLEN + s) * 256;
    const float* hb = hp + b * 256;
    float sum = 0.0f;
    for (int a = 0; a < 256; a += 4) {
      float4 e = *(const float4*)(ep + a);
      sum += tanhf(hb[a] + e.x) + tanhf(hb[a + 1] + e.y) +
             tanhf(hb[a + 2] + e.z) + tanhf(hb[a + 3] + e.w);
    }
    sc[i] = sum;
  }
  __syncthreads();
  // P2: softmax over s (one thread per batch row)
  if (tid < BSZ) {
    float mx = -3.4e38f;
    for (int s = 0; s < SLEN; ++s) mx = fmaxf(mx, sc[tid * 64 + s]);
    float sm = 0.0f;
    for (int s = 0; s < SLEN; ++s) {
      float e = __expf(sc[tid * 64 + s] - mx);
      wg[tid * 64 + s] = e; sm += e;
    }
    float inv = 1.0f / sm;
    for (int s = 0; s < SLEN; ++s) wg[tid * 64 + s] *= inv;
  }
  __syncthreads();
  // P3: ctx = sum_s w[b,s]*enc[b,s,:]; assemble xd=[embt|ctx], xout parts
  for (int i = tid; i < BSZ * 512; i += 512) {
    int b = i >> 9, d = i & 511;
    const bf16* eb = enc_bf + (size_t)b * (SLEN * 512) + d;
    const float* wb = wg + b * 64;
    float sum = 0.0f;
    for (int s = 0; s < SLEN; ++s) sum += wb[s] * bf2f(eb[s * 512]);
    bf16 v = f2bf(sum);
    xd[b * KD + 300 + d] = v;
    xout[b * KO + 256 + d] = v;
  }
  for (int i = tid; i < BSZ * EDIM; i += 512) {
    int b = i / EDIM, k = i - b * EDIM;
    bf16 v = embt[b * EPAD + k];
    xd[b * KD + k] = v;
    xout[b * KO + 768 + k] = v;
  }
  for (int i = tid; i < BSZ * (KD - 812); i += 512) {
    int b = i / (KD - 812), k = i - b * (KD - 812);
    xd[b * KD + 812 + k] = f2bf(0.0f);
  }
  __syncthreads();
  // P4a: G = xd @ W_ih_d^T + b_ih_d   (K=832 -> 26 k-steps)
  for (int t = wid; t < 192; t += 16) {
    int mt = t & 3, nt = t >> 2;
    v8f acc = {};
    for (int k = 0; k < 26; ++k)
      acc = wmma_bf(frag_ld_a(xd, KD, mt * 16, k * 32),
                    frag_ld_b(Wihd, KD, nt * 16, k * 32), acc);
    frag_st(G, 768, mt * 16, nt * 16, acc, bihd[nt * 16 + (lane & 15)]);
  }
  __syncthreads();
  // P4b: gh = h @ W_hh_d^T + b_hh_d ; r,z parts accumulate into G, n -> Gn
  for (int t = wid; t < 192; t += 16) {
    int mt = t & 3, nt = t >> 2;
    v8f acc = {};
#pragma unroll
    for (int k = 0; k < 8; ++k)
      acc = wmma_bf(frag_ld_a(hbf, UDIM, mt * 16, k * 32),
                    frag_ld_b(Whhd, UDIM, nt * 16, k * 32), acc);
    float bv = bhhd[nt * 16 + (lane & 15)];
    if (nt < 32) frag_acc(G, 768, mt * 16, nt * 16, acc, bv);
    else         frag_st(Gn, UDIM, mt * 16, (nt - 32) * 16, acc, bv);
  }
  __syncthreads();
  // P5: GRU update -> new h (f32 + bf16) and h2 slice of x_out
  for (int i = tid; i < BSZ * UDIM; i += 512) {
    int b = i >> 8, u = i & 255;
    float* Gb = G + b * 768;
    float r = sigm(Gb[u]);
    float z = sigm(Gb[256 + u]);
    float nn = tanhf(Gb[512 + u] + r * Gn[b * UDIM + u]);
    float hnew = (1.0f - z) * nn + z * hf32[i];
    hf32[i] = hnew;
    bf16 hv = f2bf(hnew);
    hbf[i] = hv;
    xout[b * KO + u] = hv;
  }
}

// ---------------------------------------------------------------------------
// Logits: out[64,32000] = x_out[64,1088] @ W_out[32000,1088]^T + b_out.
// X staged in LDS (139 KB) via the Tensor Data Mover (one 2D descriptor per
// workgroup, issued by wave 0, tracked with TENSORcnt); WMMA A-fragments then
// come from ds_load_b128, B-fragments stream from L2-resident bf16 W_out.
// ---------------------------------------------------------------------------
__global__ __launch_bounds__(256) void k_logits(const bf16* __restrict__ xout,
                                                const bf16* __restrict__ Wout,
                                                const float* __restrict__ bout,
                                                float* __restrict__ out) {
  __shared__ __align__(16) bf16 X[BSZ * KO];
#if defined(__has_builtin) && __has_builtin(__builtin_amdgcn_tensor_load_to_lds)
  if (threadIdx.x < 32) {           // wave 0 issues the TDM descriptor
    uint64_t ga = (uint64_t)(uintptr_t)xout;
    uint32_t la = (uint32_t)(uintptr_t)(void*)X;   // LDS byte offset (addr[31:0])
    u32x4 g0;
    g0[0] = 1u;                                    // count=1, gather off
    g0[1] = la;                                    // lds_addr
    g0[2] = (uint32_t)ga;                          // global_addr[31:0]
    g0[3] = (uint32_t)(ga >> 32) | (2u << 30);     // global_addr msbs | type=2
    i32x8 g1;
    g1[0] = (1 << 16);          // workgroup_mask=0, data_size=1 (2 bytes)
    g1[1] = (KO << 16);         // tensor_dim0 = 1088 (bits 79:48, low half)
    g1[2] = (BSZ << 16);        // tensor_dim0 hi=0 | tensor_dim1 = 64
    g1[3] = (KO << 16);         // tensor_dim1 hi=0 | tile_dim0 = 1088
    g1[4] = BSZ;                // tile_dim1 = 64, tile_dim2 = 0
    g1[5] = KO;                 // tensor_dim0_stride = 1088 (contiguous rows)
    g1[6] = 0;
    g1[7] = 0;
    i32x4 z4 = {0, 0, 0, 0};
#if __clang_major__ >= 23
    i32x8 z8 = {0, 0, 0, 0, 0, 0, 0, 0};
    __builtin_amdgcn_tensor_load_to_lds(g0, g1, z4, z4, z8, 0);
#else
    __builtin_amdgcn_tensor_load_to_lds(g0, g1, z4, z4, 0);
#endif
    __builtin_amdgcn_s_wait_tensorcnt(0);
  }
  __syncthreads();
#else
  {
    const uint4* s = (const uint4*)xout;
    uint4* d = (uint4*)X;
    for (int i = threadIdx.x; i < (BSZ * KO * 2) / 16; i += 256) d[i] = s[i];
  }
  __syncthreads();
#endif
  int wid = threadIdx.x >> 5, lane = threadIdx.x & 31;
  int n0 = (blockIdx.x * 8 + wid) * 16;   // 2000 n-tiles = 250 blocks * 8 waves
  v8f a0 = {}, a1 = {}, a2 = {}, a3 = {};
  for (int k = 0; k < KO / 32; ++k) {
    v16bf b = frag_ld_b(Wout, KO, n0, k * 32);
    if (k + 1 < KO / 32)
      __builtin_prefetch(Wout + (size_t)(n0 + (lane & 15)) * KO + (k + 1) * 32, 0, 0);
    a0 = wmma_bf(frag_ld_a(X, KO, 0,  k * 32), b, a0);
    a1 = wmma_bf(frag_ld_a(X, KO, 16, k * 32), b, a1);
    a2 = wmma_bf(frag_ld_a(X, KO, 32, k * 32), b, a2);
    a3 = wmma_bf(frag_ld_a(X, KO, 48, k * 32), b, a3);
  }
  int n = n0 + (lane & 15);
  float bv = bout[n];
  int mb = (lane >> 4) << 3;
#pragma unroll
  for (int v = 0; v < 8; ++v) {
    out[(size_t)(mb + v)      * VOCAB + n] = a0[v] + bv;
    out[(size_t)(mb + v + 16) * VOCAB + n] = a1[v] + bv;
    out[(size_t)(mb + v + 32) * VOCAB + n] = a2[v] + bv;
    out[(size_t)(mb + v + 48) * VOCAB + n] = a3[v] + bv;
  }
}

// ---------------------------------------------------------------------------
extern "C" void kernel_launch(void* const* d_in, const int* in_sizes, int n_in,
                              void* d_out, int out_size, void* d_ws, size_t ws_size,
                              hipStream_t stream) {
  (void)in_sizes; (void)n_in; (void)out_size; (void)ws_size;
  const int*   source   = (const int*)  d_in[0];
  const int*   target   = (const int*)  d_in[1];
  const float* emb_src  = (const float*)d_in[2];
  const float* emb_trg  = (const float*)d_in[3];
  const float* W_ih_f   = (const float*)d_in[4];
  const float* W_hh_f   = (const float*)d_in[5];
  const float* b_ih_f   = (const float*)d_in[6];
  const float* b_hh_f   = (const float*)d_in[7];
  const float* W_ih_b   = (const float*)d_in[8];
  const float* W_hh_b   = (const float*)d_in[9];
  const float* b_ih_b   = (const float*)d_in[10];
  const float* b_hh_b   = (const float*)d_in[11];
  const float* W_enc_fc = (const float*)d_in[12];
  const float* b_enc_fc = (const float*)d_in[13];
  const float* W_attn   = (const float*)d_in[14];
  const float* b_attn   = (const float*)d_in[15];
  const float* W_ih_d   = (const float*)d_in[16];
  const float* W_hh_d   = (const float*)d_in[17];
  const float* b_ih_d   = (const float*)d_in[18];
  const float* b_hh_d   = (const float*)d_in[19];
  const float* W_out    = (const float*)d_in[20];
  const float* b_out    = (const float*)d_in[21];
  float* out = (float*)d_out;

  size_t off = 0;
  auto carve = [&](size_t bytes) -> void* {
    void* r = (char*)d_ws + off;
    off += (bytes + 255) & ~(size_t)255;
    return r;
  };
  bf16* Wihf  = (bf16*)carve(768 * 320 * 2);
  bf16* Whhf  = (bf16*)carve(768 * 256 * 2);
  bf16* Wihb  = (bf16*)carve(768 * 320 * 2);
  bf16* Whhb  = (bf16*)carve(768 * 256 * 2);
  bf16* Wah   = (bf16*)carve(256 * 256 * 2);
  bf16* Wae   = (bf16*)carve(256 * 512 * 2);
  bf16* Wihd  = (bf16*)carve(768 * (size_t)KD * 2);
  bf16* Whhd  = (bf16*)carve(768 * 256 * 2);
  bf16* Wfc   = (bf16*)carve(256 * 512 * 2);
  bf16* Woutb = (bf16*)carve((size_t)VOCAB * KO * 2);
  bf16* embx  = (bf16*)carve((size_t)SLEN * BSZ * EPAD * 2);
  bf16* embt  = (bf16*)carve((size_t)(TLEN - 1) * BSZ * EPAD * 2);
  float* gif  = (float*)carve((size_t)SLEN * BSZ * 768 * 4);
  float* gib  = (float*)carve((size_t)SLEN * BSZ * 768 * 4);
  bf16* encbf = (bf16*)carve((size_t)BSZ * SLEN * 512 * 2);
  bf16* hfin  = (bf16*)carve(BSZ * 512 * 2);
  float* encp = (float*)carve((size_t)BSZ * SLEN * 256 * 4);
  float* hf32 = (float*)carve(BSZ * UDIM * 4);
  bf16* hbf   = (bf16*)carve(BSZ * UDIM * 2);
  bf16* xoutb = (bf16*)carve(BSZ * (size_t)KO * 2);

  auto cvt = [&](bf16* dst, const float* src, int N, int srcld, int col0, int K,
                 int Kp) {
    int total = N * Kp;
    k_cvt<<<(total + 255) / 256, 256, 0, stream>>>(dst, src, N, srcld, col0, K, Kp);
  };
  cvt(Wihf, W_ih_f, 768, 300, 0, 300, 320);
  cvt(Whhf, W_hh_f, 768, 256, 0, 256, 256);
  cvt(Wihb, W_ih_b, 768, 300, 0, 300, 320);
  cvt(Whhb, W_hh_b, 768, 256, 0, 256, 256);
  cvt(Wah,  W_attn, 256, 768, 0, 256, 256);
  cvt(Wae,  W_attn, 256, 768, 256, 512, 512);
  cvt(Wihd, W_ih_d, 768, 812, 0, 812, KD);
  cvt(Whhd, W_hh_d, 768, 256, 0, 256, 256);
  cvt(Wfc,  W_enc_fc, 256, 512, 0, 512, 512);
  cvt(Woutb, W_out, VOCAB, 1068, 0, 1068, KO);

  k_gather<<<(SLEN * BSZ * EPAD + 255) / 256, 256, 0, stream>>>(
      embx, emb_src, source, SLEN * BSZ, EDIM, EPAD);
  k_gather<<<((TLEN - 1) * BSZ * EPAD + 255) / 256, 256, 0, stream>>>(
      embt, emb_trg, target, (TLEN - 1) * BSZ, EDIM, EPAD);

  // Batched encoder input projections: [4096,320] @ [320,768]
  k_gemm<<<1536, 256, 0, stream>>>(embx, EPAD, Wihf, EPAD, b_ih_f, gif, 768,
                                   256, 48, 10);
  k_gemm<<<1536, 256, 0, stream>>>(embx, EPAD, Wihb, EPAD, b_ih_b, gib, 768,
                                   256, 48, 10);
  // Sequential bi-GRU encoder
  k_enc_gru<<<1, 512, 0, stream>>>(gif, Whhf, b_hh_f, encbf, hfin, 0);
  k_enc_gru<<<1, 512, 0, stream>>>(gib, Whhb, b_hh_b, encbf, hfin, 1);
  // Decoder initial state
  k_state<<<1, 256, 0, stream>>>(hfin, Wfc, b_enc_fc, hf32, hbf);
  // enc_proj = enc @ W_attn_e^T + b_attn : [4096,512] @ [512,256]
  k_gemm<<<512, 256, 0, stream>>>(encbf, 512, Wae, 512, b_attn, encp, 256,
                                  256, 16, 16);
  k_init<<<(BSZ * VOCAB + 255) / 256, 256, 0, stream>>>(out, xoutb);

  for (int t = 0; t < TLEN - 1; ++t) {
    k_dec_step<<<1, 512, 0, stream>>>(hf32, hbf, Wah, encp, encbf,
                                      embt + (size_t)t * BSZ * EPAD,
                                      Wihd, b_ih_d, Whhd, b_hh_d, xoutb);
    k_logits<<<250, 256, 0, stream>>>(xoutb, Woutb, b_out,
                                      out + (size_t)(t + 1) * BSZ * VOCAB);
  }
}
